// FractionalFusionUp_53584011985289
// MI455X (gfx1250) — matrix operand order repaired and checked by reading, not compiled, not run on hardware
//
#include <hip/hip_runtime.h>

// ---------------- types ----------------
typedef _Float16 h8  __attribute__((ext_vector_type(8)));
typedef _Float16 h16 __attribute__((ext_vector_type(16)));
typedef float    f8  __attribute__((ext_vector_type(8)));
typedef unsigned int u32x4 __attribute__((ext_vector_type(4)));
typedef int          i32x4 __attribute__((ext_vector_type(4)));
typedef int          i32x8 __attribute__((ext_vector_type(8)));

union H16 { h16 v; h8 h[2]; };

// ---------------- problem constants ----------------
#define NPIX 16384            // B*H*W = 4*64*64
#define K1   576              // C*9
#define N1   128              // 64 offset + 64 gate hidden channels
#define K2   1152             // 128*9 (cross-branch zero padded)
#define N2   16               // 8 offset heads + 1 gate head + 7 pad

// ws byte offsets (all 256-aligned)
#define OFF_W1   0u           // 128*576*2  = 147456
#define OFF_W2   147456u      // 16*1152*2  =  36864
#define OFF_H1   184320u      // 16384*128*2= 4194304
#define OFF_G    4378624u     // 16384*4    =   65536
#define OFF_ML   4444160u     // 16384*25*4 = 1638400
#define OFF_KH   6082560u     // 16384*25*4 = 1638400  (end 7720960)

#define YOFF_V   4194304      // d_out offset of v    [4,8,64,64]
#define YOFF_G   4325376      // d_out offset of g_up [4,1,128,128]

// ---------------- weight repack to f16 GEMM layout ----------------
__global__ __launch_bounds__(256) void pack_weights(
    const float* __restrict__ ow1, const float* __restrict__ gw1,
    const float* __restrict__ ow2, const float* __restrict__ gw2,
    _Float16* __restrict__ W1, _Float16* __restrict__ W2)
{
  int gid = blockIdx.x * 256 + threadIdx.x;
  const int T1 = N1 * K1;               // 73728
  if (gid < T1) {
    int n = gid / K1, k = gid % K1;
    int tap = k >> 6, c = k & 63;
    int ky = tap / 3, kx = tap % 3;
    const float* w = (n < 64) ? ow1 : gw1;
    int o = n & 63;
    W1[n * K1 + k] = (_Float16)w[((o * 64 + c) * 3 + ky) * 3 + kx];
  } else if (gid < T1 + N2 * K2) {      // + 18432
    int g2 = gid - T1;
    int n = g2 / K2, k = g2 % K2;
    int tap = k >> 7, cc = k & 127;
    int ky = tap / 3, kx = tap % 3;
    float val = 0.f;
    if (n < 8) {
      if (cc < 64) val = ow2[((n * 64 + cc) * 3 + ky) * 3 + kx];
    } else if (n == 8) {
      if (cc >= 64) val = gw2[((cc - 64) * 3 + ky) * 3 + kx];
    }
    W2[n * K2 + k] = (_Float16)val;
  }
}

// ---------------- conv1: implicit GEMM, WMMA f16 ----------------
// M=16384 pixels (64/block -> 4 M-subtiles per wave), N=128 (16/wave * 8 waves),
// K=576 (18 chunks of 32). Each B-fragment feeds 4 wmma.
__global__ __launch_bounds__(256) void conv1_wmma(
    const float* __restrict__ x, const _Float16* __restrict__ W1,
    const float* __restrict__ ob1, const float* __restrict__ gb1,
    _Float16* __restrict__ h1)
{
  __shared__ __align__(16) _Float16 tA[64 * 32];
  const int tid = threadIdx.x, wave = tid >> 5, lane = tid & 31;
  const int r = lane & 15, hi = lane >> 4;
  const int blk = blockIdx.x;                     // pixel tile [blk*64, +64)

  f8 acc[4];
  #pragma unroll
  for (int s = 0; s < 4; ++s)
    #pragma unroll
    for (int q = 0; q < 8; ++q) acc[s][q] = 0.f;

  // staging: thread owns pixel row m, 8-channel group jg
  const int m = tid >> 2, jg = (tid & 3) * 8;
  const int gp = blk * 64 + m;
  const int b = gp >> 12, y = (gp >> 6) & 63, xw = gp & 63;

  for (int t = 0; t < 18; ++t) {
    __syncthreads();
    const int tap = t >> 1, c0 = (t & 1) * 32;
    const int ys = y + tap / 3 - 1, xs = xw + tap % 3 - 1;
    const bool inb = ((unsigned)ys < 64u) & ((unsigned)xs < 64u);
    const float* src = x + ((b * 64 + (c0 + jg)) << 12) + (ys << 6) + xs;
    h8 hv;
    #pragma unroll
    for (int q = 0; q < 8; ++q)
      hv[q] = (_Float16)(inb ? src[q << 12] : 0.f);  // channel stride 4096 f32
    *(h8*)&tA[(m << 5) + jg] = hv;
    __syncthreads();

    // B fragment: wave's 16 output channels (reused by 4 subtiles)
    H16 bf;
    const int n = (wave << 4) + r;
    const _Float16* wp = W1 + n * K1 + t * 32 + hi * 8;
    bf.h[0] = *(const h8*)wp;
    bf.h[1] = *(const h8*)(wp + 16);
    #pragma unroll
    for (int s = 0; s < 4; ++s) {
      H16 a;
      const int row = (s << 4) + r;
      a.h[0] = *(const h8*)&tA[(row << 5) + hi * 8];
      a.h[1] = *(const h8*)&tA[(row << 5) + 16 + hi * 8];
      acc[s] = __builtin_amdgcn_wmma_f32_16x16x32_f16(
          false, a.v, false, bf.v, (short)0, acc[s], false, false);
    }
  }

  // epilogue: lane holds N = wave*16+r; rows M = s*16 + vv + 8*hi; bias+relu
  const int n = (wave << 4) + r;
  const float bias = (n < 64) ? ob1[n] : gb1[n - 64];
  #pragma unroll
  for (int s = 0; s < 4; ++s)
    #pragma unroll
    for (int vv = 0; vv < 8; ++vv) {
      const int gpo = blk * 64 + (s << 4) + vv + 8 * hi;
      float o = acc[s][vv] + bias;
      h1[gpo * 128 + n] = (_Float16)(o > 0.f ? o : 0.f);
    }
}

// ---------------- conv2: implicit GEMM, WMMA f16, TDM-staged A ----------------
// M=16384 (128/block: 16/wave * 8 waves), N=16, K=1152 (36 chunks of 32).
// A-tile (128 rows x 64B of f16 h1) is DMA'd LDS-ward by the Tensor Data Mover;
// image-border rows are then zero-patched in LDS (SAME padding).
__global__ __launch_bounds__(256) void conv2_wmma(
    const _Float16* __restrict__ h1, const _Float16* __restrict__ W2,
    const float* __restrict__ ob2, const float* __restrict__ gb2,
    float* __restrict__ out_v, float* __restrict__ g_buf)
{
  __shared__ __align__(16) _Float16 tA[128 * 32];
  const int tid = threadIdx.x, wave = tid >> 5, lane = tid & 31;
  const int r = lane & 15, hi = lane >> 4;
  const int blk = blockIdx.x;                     // pixels [blk*128, +128)

  f8 acc;
  #pragma unroll
  for (int q = 0; q < 8; ++q) acc[q] = 0.f;

  const int m = tid >> 1, sub = tid & 1;          // 2 threads per A row (patching)
  const int gp = blk * 128 + m;
  const int b = gp >> 12, y = (gp >> 6) & 63, xw = gp & 63;

  const unsigned ldsA = (unsigned)(size_t)(void*)&tA[0];   // LDS byte offset
  const unsigned long long h1base = (unsigned long long)(size_t)h1;

  for (int t = 0; t < 36; ++t) {
    __syncthreads();                              // prev-iter fragment reads done
    const int tap = t >> 2, c0 = (t & 3) * 32;
    const int dy = tap / 3 - 1, dx = tap % 3 - 1;

    if (wave == 0) {
      // D#: 2D tile, data_size=2B, tile 32 x 128, row stride 128 elements.
      const long long srcRow = (long long)(blk * 128 + dy * 64 + dx);
      const unsigned long long ga =
          h1base + (unsigned long long)(srcRow * 256 + c0 * 2);
      u32x4 g0;
      g0[0] = 1u;                                        // count=1, user mode
      g0[1] = ldsA;                                      // lds_addr (bytes)
      g0[2] = (unsigned)(ga & 0xFFFFFFFFull);            // global_addr[31:0]
      g0[3] = (unsigned)((ga >> 32) & 0x01FFFFFFull)     // global_addr[56:32]
            | 0x80000000u;                               // type=2 ("image")
      i32x8 g1;
      g1[0] = 0x00010000;                                // data_size=1 (2B)
      g1[1] = 32 << 16;                                  // tensor_dim0 = 32
      g1[2] = 128 << 16;                                 // tensor_dim1 = 128
      g1[3] = 32 << 16;                                  // tile_dim0 = 32
      g1[4] = 128;                                       // tile_dim1 = 128
      g1[5] = 128;                                       // dim0_stride = 128 elem
      g1[6] = 0; g1[7] = 0;
      i32x4 gz4; gz4[0] = gz4[1] = gz4[2] = gz4[3] = 0;
#if __clang_major__ >= 23
      i32x8 gz8;
      #pragma unroll
      for (int q = 0; q < 8; ++q) gz8[q] = 0;
      __builtin_amdgcn_tensor_load_to_lds(g0, g1, gz4, gz4, gz8, 0);
#else
      __builtin_amdgcn_tensor_load_to_lds(g0, g1, gz4, gz4, 0);
#endif
      __builtin_amdgcn_s_wait_tensorcnt(0);
    }
    __syncthreads();                              // DMA'd tile visible

    // zero-patch rows whose source pixel is outside the image (SAME padding)
    {
      const int ys = y + dy, xs = xw + dx;
      if (!(((unsigned)ys < 64u) & ((unsigned)xs < 64u))) {
        h8 z;
        #pragma unroll
        for (int q = 0; q < 8; ++q) z[q] = (_Float16)0.f;
        *(h8*)&tA[(m << 5) + sub * 16]     = z;
        *(h8*)&tA[(m << 5) + sub * 16 + 8] = z;
      }
    }
    __syncthreads();

    H16 a, bf;
    const int row = (wave << 4) + r;
    a.h[0] = *(const h8*)&tA[(row << 5) + hi * 8];
    a.h[1] = *(const h8*)&tA[(row << 5) + 16 + hi * 8];
    const _Float16* wp = W2 + r * K2 + t * 32 + hi * 8;
    bf.h[0] = *(const h8*)wp;
    bf.h[1] = *(const h8*)(wp + 16);
    acc = __builtin_amdgcn_wmma_f32_16x16x32_f16(
        false, a.v, false, bf.v, (short)0, acc, false, false);
  }

  // epilogue: n=r, pixel = blk*128 + wave*16 + vv + 8*hi
  const int n = r;
  #pragma unroll
  for (int vv = 0; vv < 8; ++vv) {
    const int gpo = blk * 128 + (wave << 4) + vv + 8 * hi;
    const int bo = gpo >> 12, yo = (gpo >> 6) & 63, xo = gpo & 63;
    if (n < 8) {
      float raw = acc[vv] + ob2[n];
      out_v[((bo * 8 + n) << 12) + (yo << 6) + xo] = 1.2f * tanhf(raw);
    } else if (n == 8) {
      float raw = acc[vv] + gb2[0];
      g_buf[gpo] = 1.f / (1.f + expf(-raw));
    }
  }
}

// ---------------- per-pixel fractional masks ----------------
__global__ __launch_bounds__(256) void mask_kernel(
    const float* __restrict__ out_v, float* __restrict__ mask_lp,
    float* __restrict__ ker_hp)
{
  const int p = blockIdx.x * 256 + threadIdx.x;   // 16384
  const int b = p >> 12, y = (p >> 6) & 63, xw = p & 63;

  float w1[8], w2[8];
  #pragma unroll
  for (int d = 0; d < 8; ++d) {
    float v = out_v[((b * 8 + d) << 12) + (y << 6) + xw];
    w1[d] = -v;                    // m=1 GL coeff
    w2[d] = 0.5f * v * (v - 1.f);  // m=2 GL coeff
  }

  const float R = 0.70710678f;
  const float dxs[8] = {1.f,  R, 0.f, -R, -1.f, -R, 0.f,  R};
  const float dys[8] = {0.f,  R, 1.f,  R,  0.f, -R, -1.f, -R};

  float mp[25];
  #pragma unroll
  for (int t = 0; t < 25; ++t) {
    const float ky = (float)(t / 5 - 2), kx = (float)(t % 5 - 2);
    float s = 0.f;
    #pragma unroll
    for (int d = 0; d < 8; ++d) {
      float proj = ky * dys[d] + kx * dxs[d];
      float st = rintf(proj);                    // round half-even == np.round
      st = fminf(2.f, fmaxf(-2.f, st));
      int si = (int)st;
      s += w1[d] * (float)((si == 1) - (si == -1))
         + w2[d] * (float)((si == 2) - (si == -2));
    }
    s *= 0.125f;
    if (t == 12) s += 2.0f;        // center bias (pre-softmax, shared w/ hp)
    mp[t] = s;
  }

  float mx = -1e30f;
  #pragma unroll
  for (int t = 0; t < 25; ++t) mx = fmaxf(mx, mp[t]);
  float e[25], se = 0.f, mean = 0.f;
  #pragma unroll
  for (int t = 0; t < 25; ++t) {
    e[t] = expf((mp[t] - mx) * (1.f / 0.6f));
    se += e[t];
    mean += mp[t];
  }
  mean *= (1.f / 25.f);
  float sa = 0.f;
  #pragma unroll
  for (int t = 0; t < 25; ++t) sa += fabsf(mp[t] - mean);
  const float invs = 1.f / se;
  const float invh = 1.f / (sa + 1e-8f);
  #pragma unroll
  for (int t = 0; t < 25; ++t) {
    mask_lp[p * 25 + t] = e[t] * invs;
    ker_hp[p * 25 + t]  = (mp[t] - mean) * invh;
  }
}

// ---------------- gather + upsample + blend ----------------
__global__ __launch_bounds__(256) void fuse_kernel(
    const float* __restrict__ x, const float* __restrict__ mask_lp,
    const float* __restrict__ ker_hp, const float* __restrict__ g_buf,
    const float* __restrict__ beta, float* __restrict__ y_out,
    float* __restrict__ gup_out)
{
  const int gid = blockIdx.x * 256 + threadIdx.x;  // 1048576
  const int p = gid >> 6, c = gid & 63;
  const int b = p >> 12, y = (p >> 6) & 63, xw = p & 63;
  const float* xc = x + ((b * 64 + c) << 12);

  const float alp = tanhf(beta[0]);
  const float g = g_buf[p];
  const float lam = 0.2f * (1.f - g);

  const float* ml = mask_lp + p * 25;
  const float* kh = ker_hp + p * 25;
  float acc_lp = 0.f, acc_hp = 0.f;
  #pragma unroll
  for (int t = 0; t < 25; ++t) {
    const int ky = t / 5 - 2, kx = t % 5 - 2;
    int ys = y + ky;  ys = ys < 0 ? -ys : (ys > 63 ? 126 - ys : ys);   // reflect
    int xs = xw + kx; xs = xs < 0 ? -xs : (xs > 63 ? 126 - xs : xs);
    const float xv = xc[(ys << 6) + xs];
    acc_lp += xv * ml[t];
    acc_hp += xv * kh[t];
  }
  const float hpterm = alp * g * acc_hp;

  // half-pixel bilinear x2 upsample of x (edge clamp)
  const float x00 = xc[(y << 6) + xw];
  const int ym = y > 0 ? y - 1 : 0, yp = y < 63 ? y + 1 : 63;
  const int xm = xw > 0 ? xw - 1 : 0, xp = xw < 63 ? xw + 1 : 63;
  #pragma unroll
  for (int i = 0; i < 2; ++i) {
    const int yn = i ? yp : ym;
    #pragma unroll
    for (int j = 0; j < 2; ++j) {
      const int xn = j ? xp : xm;
      const float xup = 0.5625f * x00
                      + 0.1875f * xc[(yn << 6) + xw]
                      + 0.1875f * xc[(y << 6) + xn]
                      + 0.0625f * xc[(yn << 6) + xn];
      const float ylp = xup + lam * (acc_lp - xup);
      const int Y = 2 * y + i, X = 2 * xw + j;
      y_out[((b * 64 + c) << 14) + (Y << 7) + X] = ylp + hpterm;
    }
  }
  if (c == 0) {
    #pragma unroll
    for (int i = 0; i < 2; ++i)
      #pragma unroll
      for (int j = 0; j < 2; ++j)
        gup_out[(b << 14) + ((2 * y + i) << 7) + (2 * xw + j)] = g;
  }
}

// ---------------- launcher ----------------
extern "C" void kernel_launch(void* const* d_in, const int* in_sizes, int n_in,
                              void* d_out, int out_size, void* d_ws, size_t ws_size,
                              hipStream_t stream) {
  const float* x    = (const float*)d_in[0];
  const float* ow1  = (const float*)d_in[1];
  const float* ob1  = (const float*)d_in[2];
  const float* ow2  = (const float*)d_in[3];
  const float* ob2  = (const float*)d_in[4];
  const float* gw1  = (const float*)d_in[5];
  const float* gb1  = (const float*)d_in[6];
  const float* gw2  = (const float*)d_in[7];
  const float* gb2  = (const float*)d_in[8];
  const float* beta = (const float*)d_in[9];

  float* out = (float*)d_out;
  char*  ws  = (char*)d_ws;
  _Float16* W1 = (_Float16*)(ws + OFF_W1);
  _Float16* W2 = (_Float16*)(ws + OFF_W2);
  _Float16* h1 = (_Float16*)(ws + OFF_H1);
  float* g_buf   = (float*)(ws + OFF_G);
  float* mask_lp = (float*)(ws + OFF_ML);
  float* ker_hp  = (float*)(ws + OFF_KH);
  float* v_out   = out + YOFF_V;
  float* gup_out = out + YOFF_G;

  pack_weights<<<(N1 * K1 + N2 * K2 + 255) / 256, 256, 0, stream>>>(
      ow1, gw1, ow2, gw2, W1, W2);
  conv1_wmma<<<NPIX / 64, 256, 0, stream>>>(x, W1, ob1, gb1, h1);
  conv2_wmma<<<NPIX / 128, 256, 0, stream>>>(h1, W2, ob2, gb2, v_out, g_buf);
  mask_kernel<<<NPIX / 256, 256, 0, stream>>>(v_out, mask_lp, ker_hp);
  fuse_kernel<<<(NPIX * 64) / 256, 256, 0, stream>>>(
      x, mask_lp, ker_hp, g_buf, beta, out, gup_out);
}
